// LinearAnomalyAttention_68599217652262
// MI455X (gfx1250) — matrix operand order repaired and checked by reading, not compiled, not run on hardware
//
#include <hip/hip_runtime.h>
#include <hip/hip_bf16.h>

typedef __attribute__((ext_vector_type(16))) _Float16 v16h;
typedef __attribute__((ext_vector_type(8)))  float    v8f;

#define B_ 8
#define L_ 4096
#define H_ 8
#define E_ 64
#define ACC_N (B_ * H_ * E_)   // 4096 floats per accumulator
#define LOG2E_F 1.4426950408889634f

static __device__ __forceinline__ float softplus_f(float d) {
    // numerically safe softplus; delta1 ~ 1.0 (scalar, once per thread)
    return d > 20.0f ? d : log1pf(expf(d));
}

// hardware transcendental paths (keep VALU under the 11.5us HBM floor)
static __device__ __forceinline__ float fast_exp2(float x) {
    return __builtin_amdgcn_exp2f(x);      // v_exp_f32
}
static __device__ __forceinline__ float fast_rcp(float x) {
    return __builtin_amdgcn_rcpf(x);       // v_rcp_f32 (~1 ulp)
}

// reductions across a 16-lane half-wave (masks < 16 never cross halves)
static __device__ __forceinline__ float redmax16(float v) {
    v = fmaxf(v, __shfl_xor(v, 1));
    v = fmaxf(v, __shfl_xor(v, 2));
    v = fmaxf(v, __shfl_xor(v, 4));
    v = fmaxf(v, __shfl_xor(v, 8));
    return v;
}
static __device__ __forceinline__ float redsum16(float v) {
    v += __shfl_xor(v, 1);
    v += __shfl_xor(v, 2);
    v += __shfl_xor(v, 4);
    v += __shfl_xor(v, 8);
    return v;
}

static __device__ __forceinline__ unsigned int pkh2(float a, float b) {
    union { _Float16 h[2]; unsigned int u; } t;
    t.h[0] = (_Float16)a; t.h[1] = (_Float16)b;
    return t.u;
}

// ---------------------------------------------------------------------------
// Pass 0: zero the workspace accumulators (k_sum | kv_diag), 8192 floats.
// ---------------------------------------------------------------------------
__global__ void laa_zero_ws(float* __restrict__ ws) {
    int i = blockIdx.x * 256 + threadIdx.x;
    if (i < 2 * ACC_N) ws[i] = 0.0f;
}

// ---------------------------------------------------------------------------
// Pass 1: stream keys+values once; per-row clamped softmax of K; accumulate
//   k_sum[b,h,e]   += k_hat
//   kv_diag[b,h,e] += k_hat * v
// Block = fixed b, 32 consecutive l, all h. Wave w owns h = w.
// Each wave: 16 lanes per row (float4/lane), two rows (lane halves), 16 iters.
// ---------------------------------------------------------------------------
__global__ void laa_pass1(const float* __restrict__ keys,
                          const float* __restrict__ values,
                          const float* __restrict__ delta1,
                          float* __restrict__ ws) {
    const float escale = LOG2E_F / softplus_f(delta1[0]);  // exp(x/sp)=2^(x*escale)

    const int blk   = blockIdx.x;
    const int b     = blk >> 7;          // / 128
    const int chunk = blk & 127;         // 128 chunks of 32 l
    const int tid   = threadIdx.x;
    const int h     = tid >> 5;          // wave id == head
    const int lane  = tid & 31;
    const int li    = lane & 15;
    const int half  = lane >> 4;
    const int e0    = li * 4;

    float accK[4]  = {0.f, 0.f, 0.f, 0.f};
    float accKV[4] = {0.f, 0.f, 0.f, 0.f};

    for (int it = 0; it < 16; ++it) {
        const int l = chunk * 32 + it * 2 + half;
        const size_t idx = (((size_t)b * L_ + l) * H_ + h) * E_ + e0;
        float4 kq = *(const float4*)(keys + idx);
        float4 vv = *(const float4*)(values + idx);
        if (it < 15) {  // gfx1250 global_prefetch_b8 for the next pair of rows
            __builtin_prefetch(keys + idx + 2 * H_ * E_, 0, 3);
            __builtin_prefetch(values + idx + 2 * H_ * E_, 0, 3);
        }
        float k0 = kq.x < 0.f ? -100.f : kq.x;
        float k1 = kq.y < 0.f ? -100.f : kq.y;
        float k2 = kq.z < 0.f ? -100.f : kq.z;
        float k3 = kq.w < 0.f ? -100.f : kq.w;
        float m = redmax16(fmaxf(fmaxf(k0, k1), fmaxf(k2, k3)));
        float p0 = fast_exp2((k0 - m) * escale);
        float p1 = fast_exp2((k1 - m) * escale);
        float p2 = fast_exp2((k2 - m) * escale);
        float p3 = fast_exp2((k3 - m) * escale);
        float inv = fast_rcp(redsum16(p0 + p1 + p2 + p3));  // sum >= 1
        p0 *= inv; p1 *= inv; p2 *= inv; p3 *= inv;
        accK[0] += p0; accK[1] += p1; accK[2] += p2; accK[3] += p3;
        accKV[0] += p0 * vv.x; accKV[1] += p1 * vv.y;
        accKV[2] += p2 * vv.z; accKV[3] += p3 * vv.w;
    }

    // fold the two lane halves (different l, same (h,e))
    #pragma unroll
    for (int j = 0; j < 4; ++j) {
        accK[j]  += __shfl_xor(accK[j], 16);
        accKV[j] += __shfl_xor(accKV[j], 16);
    }
    if (half == 0) {
        const int base = (b * H_ + h) * E_ + e0;
        float* ksum = ws;
        float* kvd  = ws + ACC_N;
        #pragma unroll
        for (int j = 0; j < 4; ++j) {
            atomicAdd(&ksum[base + j], accK[j]);
            atomicAdd(&kvd[base + j],  accKV[j]);
        }
    }
}

// ---------------------------------------------------------------------------
// Pass 2: stream queries once; per-row softmax; z-dot via WMMA; write output.
// Block = fixed (b,h), 16 consecutive l. 8 waves x 2 rows. Normalized q_hat
// is staged in LDS as f16 [16][64]; wave 0 computes all 16 row-dots
// D = A(16x64 q_hat, two K=32 chunks) x B(k_sum broadcast to all 16 columns)
// with two v_wmma_f32_16x16x32_f16, then z = 1/(dot + 1e-6).
// ---------------------------------------------------------------------------
__global__ void laa_pass2(const float* __restrict__ queries,
                          const float* __restrict__ delta1,
                          const float* __restrict__ ws,
                          float* __restrict__ out) {
    __shared__ unsigned int qtile[16][32];  // f16 pairs: row-major [16][64] halves
    __shared__ float zsh[16];

    const float escale = LOG2E_F / softplus_f(delta1[0]);

    const int blk   = blockIdx.x;
    const int ltile = blk & 255;
    const int h     = (blk >> 8) & 7;
    const int b     = blk >> 11;
    const int tid   = threadIdx.x;
    const int w     = tid >> 5;
    const int lane  = tid & 31;
    const int li    = lane & 15;
    const int half  = lane >> 4;
    const int r     = w * 2 + half;          // row 0..15 within tile
    const int l     = ltile * 16 + r;
    const int e0    = li * 4;

    const size_t idx = (((size_t)b * L_ + l) * H_ + h) * E_ + e0;
    float4 qq = *(const float4*)(queries + idx);
    float q0 = qq.x < 0.f ? -100.f : qq.x;
    float q1 = qq.y < 0.f ? -100.f : qq.y;
    float q2 = qq.z < 0.f ? -100.f : qq.z;
    float q3 = qq.w < 0.f ? -100.f : qq.w;
    float m = redmax16(fmaxf(fmaxf(q0, q1), fmaxf(q2, q3)));
    float p0 = fast_exp2((q0 - m) * escale);
    float p1 = fast_exp2((q1 - m) * escale);
    float p2 = fast_exp2((q2 - m) * escale);
    float p3 = fast_exp2((q3 - m) * escale);
    float inv = fast_rcp(redsum16(p0 + p1 + p2 + p3));  // sum >= 1
    p0 *= inv; p1 *= inv; p2 *= inv; p3 *= inv;

    // stage q_hat as f16 in the WMMA A staging tile
    qtile[r][2 * li]     = pkh2(p0, p1);
    qtile[r][2 * li + 1] = pkh2(p2, p3);
    __syncthreads();

    const float* ksum = ws + (b * H_ + h) * E_;

    if (w == 0) {
        // A fragment (16-bit A 16x32): lane m holds K=0..7,16..23 of row m,
        // lane m+16 holds K=8..15,24..31 (as packed pairs).
        const unsigned int* rowp = qtile[li];
        union { unsigned int u[8]; v16h v; } A0, A1, Bm0, Bm1;
        #pragma unroll
        for (int j = 0; j < 4; ++j) {
            A0.u[j]     = rowp[4 * half + j];            // K = 2j.. / 8+2j..
            A0.u[4 + j] = rowp[8 + 4 * half + j];        // K = 16+2j / 24+2j
            A1.u[j]     = rowp[16 + 4 * half + j];       // chunk 1: e 32..63
            A1.u[4 + j] = rowp[16 + 8 + 4 * half + j];
        }
        // B fragment (16-bit B 32x16): lanes 0-15 hold K=0..7,16..23,
        // lanes 16-31 hold K=8..15,24..31; every column N = k_sum (broadcast).
        const int bb = half ? 8 : 0;
        #pragma unroll
        for (int j = 0; j < 8; ++j) {
            const int kk = bb + 2 * j + ((j & 4) ? 8 : 0);
            Bm0.u[j] = pkh2(ksum[kk],      ksum[kk + 1]);
            Bm1.u[j] = pkh2(ksum[32 + kk], ksum[32 + kk + 1]);
        }
        v8f c = {};
        c = __builtin_amdgcn_wmma_f32_16x16x32_f16(false, A0.v, false, Bm0.v,
                                                   (short)0, c, false, false);
        c = __builtin_amdgcn_wmma_f32_16x16x32_f16(false, A1.v, false, Bm1.v,
                                                   (short)0, c, false, false);
        // D layout: c[v] at lane n<16 -> (M=v, N=n); lane n+16 -> (M=v+8, N=n).
        // All columns identical; lanes 0 and 16 (N=0) publish rows 0..7 / 8..15.
        if (li == 0) {
            #pragma unroll
            for (int rr = 0; rr < 8; ++rr)
                zsh[half * 8 + rr] = fast_rcp(c[rr] + 1e-6f);
        }
    }
    __syncthreads();

    const float z = zsh[r];
    const float* kvd = ws + ACC_N + (b * H_ + h) * E_;
    float4 kv4 = *(const float4*)(kvd + e0);
    float4 o;
    o.x = p0 * kv4.x * z;
    o.y = p1 * kv4.y * z;
    o.z = p2 * kv4.z * z;
    o.w = p3 * kv4.w * z;
    *(float4*)(out + idx) = o;
}

extern "C" void kernel_launch(void* const* d_in, const int* in_sizes, int n_in,
                              void* d_out, int out_size, void* d_ws, size_t ws_size,
                              hipStream_t stream) {
    (void)in_sizes; (void)n_in; (void)out_size; (void)ws_size;
    const float* queries = (const float*)d_in[0];
    const float* keys    = (const float*)d_in[1];
    const float* values  = (const float*)d_in[2];
    const float* delta1  = (const float*)d_in[3];
    float* out = (float*)d_out;
    float* ws  = (float*)d_ws;

    laa_zero_ws<<<(2 * ACC_N + 255) / 256, 256, 0, stream>>>(ws);
    laa_pass1<<<B_ * 128, 256, 0, stream>>>(keys, values, delta1, ws);
    laa_pass2<<<B_ * H_ * (L_ / 16), 256, 0, stream>>>(queries, delta1, ws, out);
}
